// Attention_63720134803723
// MI455X (gfx1250) — matrix operand order repaired and checked by reading, not compiled, not run on hardware
//
#include <hip/hip_runtime.h>
#include <hip/hip_bf16.h>
#include <math.h>

// Problem constants from the reference: x[64,1024,256], W[256,1024], b[1024], u[1024], mask[64,1024]
#define B_ 64
#define T_ 1024
#define D_ 256
#define NTILES (T_ / 16)   // 64 column tiles of the score GEMM
#define KSTEPS (D_ / 32)   // 8 K-steps of 32 for wmma_f32_16x16x32_bf16
#define MTILES 8           // 8 m-tiles (128 rows) per workgroup, one per wave

typedef __attribute__((ext_vector_type(16))) __bf16          v16bf;
typedef __attribute__((ext_vector_type(16))) unsigned short  v16u16;
typedef __attribute__((ext_vector_type(4)))  unsigned short  v4u16;
typedef __attribute__((ext_vector_type(8)))  float           v8f;

__device__ __forceinline__ unsigned short f32_to_bf16(float f) {
    unsigned int u = __builtin_bit_cast(unsigned int, f);
    u += 0x7FFFu + ((u >> 16) & 1u);           // round-to-nearest-even
    return (unsigned short)(u >> 16);
}

// CDNA5 native V_TANH_F32 (confirmed in round-4 disasm).
__device__ __forceinline__ float fast_tanh(float v) {
#if __has_builtin(__builtin_amdgcn_tanhf)
    return __builtin_amdgcn_tanhf(v);
#else
    float xx = fminf(fmaxf(v, -15.0f), 15.0f);
    float e2 = __expf(2.0f * xx);
    return (e2 - 1.0f) / (e2 + 1.0f);
#endif
}

// WGP-scope prefetch (scope 0): pulls into ALL cache levels on miss, so the
// other 7 waves sharing this B stream hit WGP$.  The __builtin_prefetch path
// lowers to SCOPE_SE which skips the WGP cache entirely (ISA §10.5).
__device__ __forceinline__ void prefetch_wgp(const void* p) {
    asm volatile("global_prefetch_b8 %0, off" : : "v"(p) : "memory");
}

// ---------------------------------------------------------------------------
// Phase 0: repack W (f32 [D,T] row-major) into bf16 B-fragments.
// Layout: Wbf[j][s][lane][e], 16 bf16 per lane (32B contiguous -> 2 b128/frag).
// B-matrix 32x16 bf16 mapping: col = lane%16, k = 32*s + (e<8 ? e : e+8) + 8*(lane/16).
// ---------------------------------------------------------------------------
__global__ __launch_bounds__(256) void prep_Wbf(const float* __restrict__ W,
                                                unsigned short* __restrict__ Wbf) {
    int idx = blockIdx.x * 256 + threadIdx.x;   // 0 .. 262143
    int e = idx & 15;
    int l = (idx >> 4) & 31;
    int s = (idx >> 9) & 7;
    int j = idx >> 12;
    int n = j * 16 + (l & 15);
    int k = s * 32 + ((l >> 4) << 3) + (e & 7) + ((e >> 3) << 4);
    Wbf[idx] = f32_to_bf16(W[k * T_ + n]);
}

// ---------------------------------------------------------------------------
// Phase 1: fused scores.  One workgroup = 128 rows (8 m-tiles).  Each wave owns
// one m-tile (A resident in 64 VGPRs) and sweeps ALL 64 n-tiles two at a time
// (two independent WMMA accumulator chains); the 8 waves walk the identical
// B-fragment address stream, so B re-reads hit WGP$, and WGP-scope prefetch
// two tiles ahead hides L2 latency behind the current 16 WMMAs.
// ---------------------------------------------------------------------------
__global__ __launch_bounds__(256)
void score_kernel(const float* __restrict__ x, const float* __restrict__ bias,
                  const float* __restrict__ u, const unsigned char* __restrict__ mask,
                  const unsigned short* __restrict__ Wbf, float* __restrict__ scores) {
    __shared__ __align__(32) unsigned short Atile[MTILES * KSTEPS * 32 * 16];  // 64 KB bf16

    const int tid = threadIdx.x;
    const long rowbase = (long)blockIdx.x * (16 * MTILES);

    // Stage x[rowbase : rowbase+128, :] into LDS as A-fragments (bf16),
    // float4 at a time.  A-matrix 16x32 bf16 mapping: row = lane%16,
    // k = 32*s + (e<8 ? e : e+8) + 8*(lane/16).  4 consecutive (4-aligned) k
    // map to 4 consecutive fragment elements of the same lane -> 8B LDS store.
    for (int i = tid; i < 16 * MTILES * (D_ / 4); i += 256) {
        int row = i >> 6;              // 0..127
        int k0  = (i & 63) << 2;       // 4-aligned column
        const float4 v = *(const float4*)(x + (rowbase + row) * (long)D_ + k0);
        int mt  = row >> 4;
        int r   = row & 15;
        int s   = k0 >> 5;
        int w32 = k0 & 31;
        int hw  = (w32 >> 3) & 1;                       // half-wave
        int e0  = (w32 & 7) | ((w32 >> 4) << 3);        // first element
        int lane = (hw << 4) | r;
        v4u16 p;
        p.x = f32_to_bf16(v.x);
        p.y = f32_to_bf16(v.y);
        p.z = f32_to_bf16(v.z);
        p.w = f32_to_bf16(v.w);
        *(v4u16*)&Atile[((mt * KSTEPS + s) * 32 + lane) * 16 + e0] = p;
    }
    __syncthreads();

    const int wave = tid >> 5;
    const int lane = tid & 31;
    const v16u16* Ap = (const v16u16*)Atile;
    const v16u16* Bp = (const v16u16*)Wbf;

    // Pull this wave's m-tile A fragments into registers (64 VGPRs).
    v16bf af[KSTEPS];
#pragma unroll
    for (int s = 0; s < KSTEPS; ++s)
        af[s] = __builtin_bit_cast(v16bf, Ap[(wave * KSTEPS + s) * 32 + lane]);

    float acc[8];
#pragma unroll
    for (int i = 0; i < 8; ++i) acc[i] = 0.0f;

    for (int j = 0; j < NTILES; j += 2) {
        // Warm WGP$ with the B fragments two tiles ahead (no LOADcnt/VGPRs).
        int jn = (j + 2) & (NTILES - 1);
#pragma unroll
        for (int s = 0; s < KSTEPS; ++s) {
            prefetch_wgp(&Bp[((jn + 0) * KSTEPS + s) * 32 + lane]);
            prefetch_wgp(&Bp[((jn + 1) * KSTEPS + s) * 32 + lane]);
        }

        // Two independent accumulator chains over n-tiles j and j+1.
        v8f c0 = {0.f, 0.f, 0.f, 0.f, 0.f, 0.f, 0.f, 0.f};
        v8f c1 = {0.f, 0.f, 0.f, 0.f, 0.f, 0.f, 0.f, 0.f};
#pragma unroll
        for (int s = 0; s < KSTEPS; ++s) {
            v16bf b0 = __builtin_bit_cast(v16bf, Bp[((j + 0) * KSTEPS + s) * 32 + lane]);
            v16bf b1 = __builtin_bit_cast(v16bf, Bp[((j + 1) * KSTEPS + s) * 32 + lane]);
            c0 = __builtin_amdgcn_wmma_f32_16x16x32_bf16(false, af[s], false, b0,
                                                         (short)0, c0, false, false);
            c1 = __builtin_amdgcn_wmma_f32_16x16x32_bf16(false, af[s], false, b1,
                                                         (short)0, c1, false, false);
        }
        // C layout: lane holds column n = 16*j + lane%16, rows i + 8*(lane/16).
        int col0 = (j + 0) * 16 + (lane & 15);
        int col1 = (j + 1) * 16 + (lane & 15);
        float bv0 = bias[col0], uv0 = u[col0];
        float bv1 = bias[col1], uv1 = u[col1];
#pragma unroll
        for (int i = 0; i < 8; ++i) {
            acc[i] += fast_tanh(c0[i] + bv0) * uv0;
            acc[i] += fast_tanh(c1[i] + bv1) * uv1;
        }
    }

    // Reduce across the 16 lanes that share a row (xor offsets stay in-half).
#pragma unroll
    for (int off = 8; off >= 1; off >>= 1) {
#pragma unroll
        for (int i = 0; i < 8; ++i)
            acc[i] += __shfl_xor(acc[i], off, 32);
    }
    // Lanes 0 and 16 hold the 16 row sums of this wave's m-tile; finalize.
    if ((lane & 15) == 0) {
        int hw = lane >> 4;
#pragma unroll
        for (int i = 0; i < 8; ++i) {
            long row = rowbase + wave * 16 + hw * 8 + i;   // row = b*T + t
            float m = (float)mask[row];
            scores[row] = __expf(acc[i]) * m;
        }
    }
}

// ---------------------------------------------------------------------------
// Phase 2: per-batch normalize (sum + EPS) and out[b,d] = sum_t x[b,t,d]*w[t].
// Memory-bound: coalesced x reads, one block per batch.
// ---------------------------------------------------------------------------
__global__ __launch_bounds__(256)
void output_kernel(const float* __restrict__ x, const float* __restrict__ scores,
                   float* __restrict__ out) {
    __shared__ float wgt[T_];
    __shared__ float red[256];
    const int tid = threadIdx.x;
    const int b = blockIdx.x;

    const float* sb = scores + (long)b * T_;
    float local = 0.f;
    for (int t = tid; t < T_; t += 256) {
        float v = sb[t];
        wgt[t] = v;
        local += v;
    }
    red[tid] = local;
    __syncthreads();
    for (int off = 128; off > 0; off >>= 1) {
        if (tid < off) red[tid] += red[tid + off];
        __syncthreads();
    }
    float inv = 1.0f / (red[0] + 1e-7f);
    for (int t = tid; t < T_; t += 256) wgt[t] *= inv;
    __syncthreads();

    const float* xb = x + (long)b * T_ * D_;
    float accv = 0.f;
#pragma unroll 4
    for (int t = 0; t < T_; ++t)
        accv = fmaf(xb[(long)t * D_ + tid], wgt[t], accv);
    out[(long)b * D_ + tid] = accv;
}

// ---------------------------------------------------------------------------
extern "C" void kernel_launch(void* const* d_in, const int* in_sizes, int n_in,
                              void* d_out, int out_size, void* d_ws, size_t ws_size,
                              hipStream_t stream) {
    (void)in_sizes; (void)n_in; (void)out_size; (void)ws_size;
    const float*         x    = (const float*)d_in[0];
    const float*         W    = (const float*)d_in[1];
    const float*         bv   = (const float*)d_in[2];
    const float*         uv   = (const float*)d_in[3];
    const unsigned char* mask = (const unsigned char*)d_in[4];
    float*               out  = (float*)d_out;

    const size_t wbf_elems = (size_t)NTILES * KSTEPS * 32 * 16;   // 262144
    unsigned short* Wbf    = (unsigned short*)d_ws;               // 512 KB
    float*          scores = (float*)((char*)d_ws + wbf_elems * sizeof(unsigned short)); // 256 KB

    prep_Wbf<<<wbf_elems / 256, 256, 0, stream>>>(W, Wbf);
    score_kernel<<<(B_ * T_) / (16 * MTILES), 256, 0, stream>>>(x, bv, uv, mask, Wbf, scores);
    output_kernel<<<B_, 256, 0, stream>>>(x, scores, out);
}
